// Decoder_6030134084197
// MI455X (gfx1250) — matrix operand order repaired and checked by reading, not compiled, not run on hardware
//
#include <hip/hip_runtime.h>
#include <hip/hip_bf16.h>
#include <math.h>

#define B_ 64
#define S_ 64
#define E_ 256
#define H_ 512
#define V_ 50000
#define VP_ 50048   // V padded to multiple of 64 for branch-free WMMA GEMM
#define T_ 32
#define SOS_ 1

typedef __bf16 bf16;
typedef __bf16 v16bf __attribute__((ext_vector_type(16)));
typedef float  v8f   __attribute__((ext_vector_type(8)));
typedef unsigned int u32x4 __attribute__((ext_vector_type(4)));

union Frag16 { v16bf v; u32x4 u[2]; };

// ---- WMMA fragment loaders (ISA 7.12.2 layouts) ------------------------------
// A (16x32 bf16, row-major [M,K]): lane r(0..15),hi: VGPR0-3 = K[hi*8 .. hi*8+7],
// VGPR4-7 = K[16+hi*8 .. 16+hi*8+7]  -> two 16B chunks at +hi*8 and +16+hi*8.
__device__ __forceinline__ v16bf load_a_frag(const bf16* base, int ld, int m0, int k0, int lane) {
    int r = lane & 15, hi = lane >> 4;
    const bf16* p = base + (size_t)(m0 + r) * ld + k0 + hi * 8;
    Frag16 f;
    f.u[0] = *(const u32x4*)p;
    f.u[1] = *(const u32x4*)(p + 16);
    return f.v;
}
// B (32x16 bf16, K x N), stored here as [N,K] row-major (weight [out,in]).
// Column-major-per-lane: lane n holds K[hi*8 .. hi*8+7] and K[16+hi*8 .. 16+hi*8+7].
__device__ __forceinline__ v16bf load_b_frag(const bf16* base, int ld, int n0, int k0, int lane) {
    int r = lane & 15, hi = lane >> 4;
    const bf16* p = base + (size_t)(n0 + r) * ld + k0 + hi * 8;
    Frag16 f;
    f.u[0] = *(const u32x4*)p;
    f.u[1] = *(const u32x4*)(p + 16);
    return f.v;
}

struct Frags { v16bf a, b0, b1, b2, b3; };

__device__ __forceinline__ Frags load_frags(const bf16* __restrict__ A,
                                            const bf16* __restrict__ Bm,
                                            int K, int m0, int n0, int k0, int lane) {
    Frags f;
    f.a  = load_a_frag(A,  K, m0,      k0, lane);
    f.b0 = load_b_frag(Bm, K, n0,      k0, lane);
    f.b1 = load_b_frag(Bm, K, n0 + 16, k0, lane);
    f.b2 = load_b_frag(Bm, K, n0 + 32, k0, lane);
    f.b3 = load_b_frag(Bm, K, n0 + 48, k0, lane);
    return f;
}

__device__ __forceinline__ void mma4(const Frags& f, v8f& a0, v8f& a1, v8f& a2, v8f& a3) {
    a0 = __builtin_amdgcn_wmma_f32_16x16x32_bf16(false, f.a, false, f.b0, (short)0, a0, false, false);
    a1 = __builtin_amdgcn_wmma_f32_16x16x32_bf16(false, f.a, false, f.b1, (short)0, a1, false, false);
    a2 = __builtin_amdgcn_wmma_f32_16x16x32_bf16(false, f.a, false, f.b2, (short)0, a2, false, false);
    a3 = __builtin_amdgcn_wmma_f32_16x16x32_bf16(false, f.a, false, f.b3, (short)0, a3, false, false);
}

// ---- Branch-free, software-pipelined bf16 WMMA GEMM --------------------------
// C[M,N] = A[M,K] * B[N,K]^T (+bias[N])
// 128 threads = 4 waves; each wave computes a full 16(M) x 64(N) strip.
// REQUIRES: M%16==0, N%64==0, K%64==0, (M/16)*(N/64) % 4 == 0 (exact grid).
// 2-deep register pipeline: loads for chunk k+32 / k+64 are issued before the
// WMMAs consuming chunk k, so the waitcnt pass can use partial s_wait_loadcnt.
__global__ void gemm_bf16_wmma(const bf16* __restrict__ A, const bf16* __restrict__ Bm,
                               float* __restrict__ C, const float* __restrict__ bias,
                               int M, int N, int K) {
    const int lane = threadIdx.x & 31;
    const int wave = threadIdx.x >> 5;
    const int tilesM = M >> 4;
    const int gw = blockIdx.x * 4 + wave;
    const int m0 = (gw % tilesM) << 4;
    const int n0 = (gw / tilesM) << 6;
    const int r  = lane & 15;
    const int hi = lane >> 4;

    v8f acc0, acc1, acc2, acc3;
#pragma unroll
    for (int i = 0; i < 8; ++i) { acc0[i] = 0.f; acc1[i] = 0.f; acc2[i] = 0.f; acc3[i] = 0.f; }

    Frags f0 = load_frags(A, Bm, K, m0, n0, 0, lane);
    int k0 = 0;
    for (; k0 + 64 < K; k0 += 64) {
        __builtin_prefetch(Bm + (size_t)(n0 + r) * K + k0 + 96, 0, 1);
        Frags f1 = load_frags(A, Bm, K, m0, n0, k0 + 32, lane);
        mma4(f0, acc0, acc1, acc2, acc3);
        f0 = load_frags(A, Bm, K, m0, n0, k0 + 64, lane);
        mma4(f1, acc0, acc1, acc2, acc3);
    }
    // Epilogue: f0 holds chunk K-64; load final chunk K-32 and finish.
    Frags f1 = load_frags(A, Bm, K, m0, n0, K - 32, lane);
    mma4(f0, acc0, acc1, acc2, acc3);
    mma4(f1, acc0, acc1, acc2, acc3);

    // C/D layout: lane(0..15)->N=lane, VGPR i -> M = hi*8 + i
    v8f* accs[4] = { &acc0, &acc1, &acc2, &acc3 };
#pragma unroll
    for (int j = 0; j < 4; ++j) {
        int n = n0 + j * 16 + r;
        float bv = bias ? bias[n] : 0.0f;
        float* cp = C + (size_t)(m0 + hi * 8) * N + n;
        v8f av = *accs[j];
#pragma unroll
        for (int i = 0; i < 8; ++i) cp[(size_t)i * N] = av[i] + bv;
    }
}

// ---- Small helper kernels ----------------------------------------------------
__global__ void f32_to_bf16_k(const float* __restrict__ in, bf16* __restrict__ out, int n) {
    int i = blockIdx.x * blockDim.x + threadIdx.x;
    if (i < n) out[i] = (bf16)in[i];
}

// Convert [rows_valid, cols] f32 -> [rows_pad, cols] bf16, zero-filled pad rows.
__global__ void convert_pad_rows_k(const float* __restrict__ in, bf16* __restrict__ out,
                                   int rows_valid, int cols, int total_pad_elems) {
    int i = blockIdx.x * blockDim.x + threadIdx.x;
    if (i >= total_pad_elems) return;
    int n = i / cols;
    out[i] = (n < rows_valid) ? (bf16)in[i] : (bf16)0.0f;
}

__global__ void pad_bias_k(const float* __restrict__ in, float* __restrict__ out,
                           int n_valid, int n_pad) {
    int i = blockIdx.x * blockDim.x + threadIdx.x;
    if (i < n_pad) out[i] = (i < n_valid) ? in[i] : 0.0f;
}

__global__ void init_state_k(const float* __restrict__ enc_hidden, float* __restrict__ h,
                             bf16* __restrict__ h_bf, int* __restrict__ dec_in, int n) {
    int i = blockIdx.x * blockDim.x + threadIdx.x;
    if (i < n) { float v = enc_hidden[i]; h[i] = v; h_bf[i] = (bf16)v; }
    if (i < B_) dec_in[i] = SOS_;
}

__global__ void embed_gather_k(const float* __restrict__ emb, const int* __restrict__ dec_in,
                               bf16* __restrict__ x) {
    int b = blockIdx.x, e = threadIdx.x;                  // 64 blocks x 256 threads
    int tok = dec_in[b];
    x[b * E_ + e] = (bf16)emb[(size_t)tok * E_ + e];
}

__global__ void gru_elem_k(const float* __restrict__ gi, const float* __restrict__ gh,
                           float* __restrict__ h, bf16* __restrict__ h_bf,
                           bf16* __restrict__ concat_bf) {
    int idx = blockIdx.x * blockDim.x + threadIdx.x;      // B*H threads
    int b = idx / H_, j = idx % H_;
    const float* gib = gi + (size_t)b * 3 * H_;
    const float* ghb = gh + (size_t)b * 3 * H_;
    float r = 1.0f / (1.0f + expf(-(gib[j] + ghb[j])));
    float z = 1.0f / (1.0f + expf(-(gib[H_ + j] + ghb[H_ + j])));
    float n = tanhf(gib[2 * H_ + j] + r * ghb[2 * H_ + j]);
    float hn = (1.0f - z) * n + z * h[idx];
    h[idx] = hn;
    h_bf[idx] = (bf16)hn;
    concat_bf[(size_t)b * 2 * H_ + j] = (bf16)hn;
}

__global__ void attn_scores_k(const float* __restrict__ h, const float* __restrict__ enc_proj,
                              float* __restrict__ scores) {
    int idx = blockIdx.x * blockDim.x + threadIdx.x;      // B*S threads
    int b = idx / S_, s = idx % S_;
    const float4* hp = (const float4*)(h + (size_t)b * H_);
    const float4* ep = (const float4*)(enc_proj + ((size_t)b * S_ + s) * H_);
    float acc = 0.0f;
    for (int i = 0; i < H_ / 4; ++i) {
        float4 a = hp[i], c = ep[i];
        acc += a.x * c.x + a.y * c.y + a.z * c.z + a.w * c.w;
    }
    scores[idx] = acc;
}

__global__ void softmax_context_k(const float* __restrict__ scores,
                                  const float* __restrict__ enc_out,
                                  bf16* __restrict__ concat_bf) {
    __shared__ float w[S_];
    int b = blockIdx.x;                                   // 64 blocks x 512 threads
    if (threadIdx.x == 0) {
        const float* sc = scores + (size_t)b * S_;
        float mx = sc[0];
        for (int s = 1; s < S_; ++s) mx = fmaxf(mx, sc[s]);
        float sum = 0.0f;
        for (int s = 0; s < S_; ++s) { float e = expf(sc[s] - mx); w[s] = e; sum += e; }
        float inv = 1.0f / sum;
        for (int s = 0; s < S_; ++s) w[s] *= inv;
    }
    __syncthreads();
    int j = threadIdx.x;
    const float* eb = enc_out + (size_t)b * S_ * H_ + j;
    float acc = 0.0f;
    for (int s = 0; s < S_; ++s) acc += w[s] * eb[(size_t)s * H_];
    concat_bf[(size_t)b * 2 * H_ + H_ + j] = (bf16)acc;
}

__global__ void tanh_bf16_k(const float* __restrict__ in, bf16* __restrict__ out, int n) {
    int i = blockIdx.x * blockDim.x + threadIdx.x;
    if (i < n) out[i] = (bf16)tanhf(in[i]);
}

// Per-batch-row log-softmax over V entries (row stride ld), write out, argmax -> dec_in.
__global__ void logsoftmax_argmax_k(const float* __restrict__ logits, int ld,
                                    float* __restrict__ out, int* __restrict__ dec_in, int t) {
    __shared__ float smax[512];
    __shared__ int   simax[512];
    __shared__ float ssum[512];
    int b = blockIdx.x, tid = threadIdx.x, nt = blockDim.x;
    const float* lb = logits + (size_t)b * ld;
    float mx = -INFINITY; int mi = 0;
    for (int v = tid; v < V_; v += nt) { float x = lb[v]; if (x > mx) { mx = x; mi = v; } }
    smax[tid] = mx; simax[tid] = mi; __syncthreads();
    for (int s = nt >> 1; s > 0; s >>= 1) {
        if (tid < s) {
            float xo = smax[tid + s]; int io = simax[tid + s];
            if (xo > smax[tid] || (xo == smax[tid] && io < simax[tid])) {
                smax[tid] = xo; simax[tid] = io;
            }
        }
        __syncthreads();
    }
    float gmax = smax[0];
    float sum = 0.0f;
    for (int v = tid; v < V_; v += nt) sum += expf(lb[v] - gmax);
    ssum[tid] = sum; __syncthreads();
    for (int s = nt >> 1; s > 0; s >>= 1) { if (tid < s) ssum[tid] += ssum[tid + s]; __syncthreads(); }
    float lse = gmax + logf(ssum[0]);
    float* ob = out + ((size_t)b * T_ + t) * V_;
    for (int v = tid; v < V_; v += nt) ob[v] = lb[v] - lse;
    if (tid == 0) dec_in[b] = simax[0];
}

__global__ void copy_f32_k(const float* __restrict__ src, float* __restrict__ dst, int n) {
    int i = blockIdx.x * blockDim.x + threadIdx.x;
    if (i < n) dst[i] = src[i];
}

// ---- Host side ---------------------------------------------------------------
static inline size_t align256(size_t x) { return (x + 255) & ~(size_t)255; }

extern "C" void kernel_launch(void* const* d_in, const int* in_sizes, int n_in,
                              void* d_out, int out_size, void* d_ws, size_t ws_size,
                              hipStream_t stream) {
    const float* enc_hidden = (const float*)d_in[0];   // [1,B,H]
    const float* enc_out    = (const float*)d_in[3];   // [B,S,H]
    const float* embedding  = (const float*)d_in[4];   // [V,E]
    const float* w_ih       = (const float*)d_in[5];   // [3H,E]
    const float* w_hh       = (const float*)d_in[6];   // [3H,H]
    const float* b_ih       = (const float*)d_in[7];   // [3H]
    const float* b_hh       = (const float*)d_in[8];   // [3H]
    const float* attn_w     = (const float*)d_in[9];   // [H,H]
    const float* Wa         = (const float*)d_in[10];  // [H,2H]
    const float* fc_w       = (const float*)d_in[11];  // [V,H]
    const float* fc_b       = (const float*)d_in[12];  // [V]
    float* out = (float*)d_out;

    // Workspace carve-up (256B aligned)
    char* ws = (char*)d_ws; size_t cur = 0;
    auto alloc = [&](size_t bytes) { void* p = ws + cur; cur = align256(cur + bytes); return p; };
    bf16*  fcw_bf   = (bf16*)alloc((size_t)VP_ * H_ * 2);      // zero-padded rows
    float* fcb_pad  = (float*)alloc((size_t)VP_ * 4);          // zero-padded bias
    bf16*  wih_bf   = (bf16*)alloc((size_t)3 * H_ * E_ * 2);
    bf16*  whh_bf   = (bf16*)alloc((size_t)3 * H_ * H_ * 2);
    bf16*  wa_bf    = (bf16*)alloc((size_t)H_ * 2 * H_ * 2);
    bf16*  attnw_bf = (bf16*)alloc((size_t)H_ * H_ * 2);
    bf16*  encb     = (bf16*)alloc((size_t)B_ * S_ * H_ * 2);
    float* enc_proj = (float*)alloc((size_t)B_ * S_ * H_ * 4);
    float* logits   = (float*)alloc((size_t)B_ * VP_ * 4);     // ld = VP_
    float* gi       = (float*)alloc((size_t)B_ * 3 * H_ * 4);
    float* gh       = (float*)alloc((size_t)B_ * 3 * H_ * 4);
    float* h        = (float*)alloc((size_t)B_ * H_ * 4);
    bf16*  h_bf     = (bf16*)alloc((size_t)B_ * H_ * 2);
    bf16*  x_bf     = (bf16*)alloc((size_t)B_ * E_ * 2);
    bf16*  cat_bf   = (bf16*)alloc((size_t)B_ * 2 * H_ * 2);
    float* co_pre   = (float*)alloc((size_t)B_ * H_ * 4);
    bf16*  co_bf    = (bf16*)alloc((size_t)B_ * H_ * 2);
    float* scores   = (float*)alloc((size_t)B_ * S_ * 4);
    int*   dec_in   = (int*)alloc((size_t)B_ * 4);
    (void)ws_size; (void)in_sizes; (void)n_in; (void)out_size;

    auto conv = [&](const float* src, bf16* dst, int n) {
        f32_to_bf16_k<<<(n + 255) / 256, 256, 0, stream>>>(src, dst, n);
    };
    // All callers satisfy: M%16==0, N%64==0, K%64==0, (M/16)*(N/64)%4==0.
    auto gemm = [&](const bf16* A, const bf16* Bm, float* C, const float* bias,
                    int M, int N, int K) {
        int tiles = (M / 16) * (N / 64);
        gemm_bf16_wmma<<<tiles / 4, 128, 0, stream>>>(A, Bm, C, bias, M, N, K);
    };

    // One-time (per call) weight / activation conversions + state init
    {
        int tot = VP_ * H_;
        convert_pad_rows_k<<<(tot + 255) / 256, 256, 0, stream>>>(fc_w, fcw_bf, V_, H_, tot);
        pad_bias_k<<<(VP_ + 255) / 256, 256, 0, stream>>>(fc_b, fcb_pad, V_, VP_);
    }
    conv(w_ih,   wih_bf,   3 * H_ * E_);
    conv(w_hh,   whh_bf,   3 * H_ * H_);
    conv(Wa,     wa_bf,    H_ * 2 * H_);
    conv(attn_w, attnw_bf, H_ * H_);
    conv(enc_out, encb,    B_ * S_ * H_);
    init_state_k<<<(B_ * H_ + 255) / 256, 256, 0, stream>>>(enc_hidden, h, h_bf, dec_in, B_ * H_);

    // enc_proj[b,s,g] = sum_h enc_out[b,s,h] * attn_w[g,h]   (A=[4096,512] x B=[512,512]^T)
    gemm(encb, attnw_bf, enc_proj, nullptr, B_ * S_, H_, H_);

    for (int t = 0; t < T_; ++t) {
        embed_gather_k<<<B_, E_, 0, stream>>>(embedding, dec_in, x_bf);
        gemm(x_bf, wih_bf, gi, b_ih, B_, 3 * H_, E_);                 // gi = x @ w_ih^T + b_ih
        gemm(h_bf, whh_bf, gh, b_hh, B_, 3 * H_, H_);                 // gh = h @ w_hh^T + b_hh
        gru_elem_k<<<(B_ * H_) / 256, 256, 0, stream>>>(gi, gh, h, h_bf, cat_bf);
        attn_scores_k<<<(B_ * S_) / 256, 256, 0, stream>>>(h, enc_proj, scores);
        softmax_context_k<<<B_, H_, 0, stream>>>(scores, enc_out, cat_bf);
        gemm(cat_bf, wa_bf, co_pre, nullptr, B_, H_, 2 * H_);         // concat @ Wa^T
        tanh_bf16_k<<<(B_ * H_) / 256, 256, 0, stream>>>(co_pre, co_bf, B_ * H_);
        gemm(co_bf, fcw_bf, logits, fcb_pad, B_, VP_, H_);            // classifier GEMM (dominant)
        logsoftmax_argmax_k<<<B_, 512, 0, stream>>>(logits, VP_, out, dec_in, t);
    }

    // h_fin -> tail of output, after [B,T,V]
    copy_f32_k<<<(B_ * H_) / 256, 256, 0, stream>>>(
        h, out + (size_t)B_ * T_ * V_, B_ * H_);
}
